// GATBody_69776038690895
// MI455X (gfx1250) — matrix operand order repaired and checked by reading, not compiled, not run on hardware
//
#include <hip/hip_runtime.h>
#include <hip/hip_bf16.h>
#include <math.h>

// ---------------------------------------------------------------------------
// Types for CDNA5 WMMA (wave32) and TDM descriptors
// ---------------------------------------------------------------------------
typedef __attribute__((ext_vector_type(16))) __bf16 v16bf;
typedef __attribute__((ext_vector_type(8)))  float  v8f;
typedef __attribute__((ext_vector_type(4)))  unsigned int u32x4;
typedef __attribute__((ext_vector_type(8)))  int i32x8;
typedef __attribute__((ext_vector_type(4)))  int i32x4;

union Frag16 { v16bf v; uint4 q[2]; };

__device__ __forceinline__ unsigned short f2bf(float f) {
  unsigned int u = __float_as_uint(f);
  unsigned int r = 0x7FFFu + ((u >> 16) & 1u);   // round-to-nearest-even
  return (unsigned short)((u + r) >> 16);
}

// ---------------------------------------------------------------------------
// Panel GEMM: C[M][Ncol] = A[M][K] @ B[K][Ncol]
//   A row-major bf16, Bt = B^T bf16 ([Ncol][K], rows contiguous)
// Block = 256 threads (8 waves). blockIdx.x -> 128 M rows, blockIdx.y -> a
// 64-column panel. The panel of Bt (64*K bf16, contiguous) is staged into LDS
// once per block by the Tensor Data Mover, then each wave computes a 16x64
// strip: 1 A fragment (global) reused across 4 WMMAs, B fragments from LDS.
// ---------------------------------------------------------------------------
template<int K>
__global__ void gemm_panel_wmma(const unsigned short* __restrict__ A,
                                const unsigned short* __restrict__ Bt,
                                float* __restrict__ C,
                                int M, int Ncol) {
  __shared__ unsigned short ldsB[64 * K];        // 16KB (K=128) / 64KB (K=512)

  const int panelN = blockIdx.y;                 // 64-column panel index
  const int blockM = blockIdx.x * 128;
  const int wv     = threadIdx.x >> 5;           // wave 0..7 (wave-uniform)
  const int lane   = threadIdx.x & 31;
  const int lm     = lane & 15;
  const int hi     = lane >> 4;

#if __has_builtin(__builtin_amdgcn_tensor_load_to_lds)
  // ---- TDM: DMA the contiguous 64*K-element Bt panel into LDS ----
  if (wv == 0) {
    const unsigned long long gaddr =
        (unsigned long long)(Bt + (size_t)panelN * 64 * K);
    const unsigned int elems = 64u * (unsigned int)K;   // <= 32768
    u32x4 g0;
    g0[0] = 1u;                                          // count=1 (valid D#)
    g0[1] = 0u;                                          // lds_addr = 0
    g0[2] = (unsigned int)gaddr;                         // global_addr[31:0]
    g0[3] = (unsigned int)((gaddr >> 32) & 0x01FFFFFFu)  // global_addr[56:32]
            | (2u << 30);                                // type=2 ("image")
    i32x8 g1;
    g1[0] = (int)(1u << 16);                 // workgroup_mask=0, data_size=2B
    g1[1] = (int)((elems & 0xFFFFu) << 16);  // tensor_dim0[15:0]
    g1[2] = (int)((elems >> 16) | (1u << 16));  // tensor_dim0[31:16] | dim1=1
    g1[3] = (int)(elems << 16);              // dim1 hi=0 | tile_dim0 = elems
    g1[4] = 0;                               // tile_dim1=0, tile_dim2=0 (1-D)
    g1[5] = (int)elems;                      // tensor_dim0_stride[31:0]
    g1[6] = 0;                               // stride0 hi | stride1 lo
    g1[7] = 0;                               // stride1 hi
    i32x4 g2 = {0, 0, 0, 0};                 // group 2 unused (<=2-D tensor)
    i32x4 g3 = {0, 0, 0, 0};                 // group 3 unused
    i32x8 g4 = {0, 0, 0, 0, 0, 0, 0, 0};     // extra group (clang-23 6-arg form)
    __builtin_amdgcn_tensor_load_to_lds(g0, g1, g2, g3, g4, 0);
    __builtin_amdgcn_s_wait_tensorcnt(0);
  }
#else
  // ---- fallback: cooperative vector copy ----
  for (int i = threadIdx.x * 8; i < 64 * K; i += 256 * 8) {
    *(uint4*)(ldsB + i) =
        *(const uint4*)(Bt + (size_t)panelN * 64 * K + i);
  }
#endif
  __syncthreads();

  // ---- compute: wave 'wv' owns rows blockM + wv*16 .. +15, all 64 cols ----
  int rowA = blockM + wv * 16 + lm;
  if (rowA > M - 1) rowA = M - 1;                // clamp loads, guard stores
  const unsigned short* Ap = A + (size_t)rowA * K;

  v8f acc[4];
  #pragma unroll
  for (int nt = 0; nt < 4; ++nt)
    acc[nt] = (v8f){0.f, 0.f, 0.f, 0.f, 0.f, 0.f, 0.f, 0.f};

  for (int k0 = 0; k0 < K; k0 += 32) {
    // A lane layout (16-bit A 16x32): lane<16 -> K {0..7,16..23},
    // lane>=16 -> K {8..15,24..31}
    Frag16 a;
    a.q[0] = *(const uint4*)(Ap + k0 +      8 * hi);
    a.q[1] = *(const uint4*)(Ap + k0 + 16 + 8 * hi);
    #pragma unroll
    for (int nt = 0; nt < 4; ++nt) {
      // B lane layout (32x16): lane n holds column n%16, lane<16 -> K {0..7,16..23},
      // lane>=16 -> K {8..15,24..31}
      const unsigned short* Bl =
          ldsB + (size_t)(nt * 16 + lm) * K + k0 + 8 * hi;
      Frag16 b;
      b.q[0] = *(const uint4*)(Bl);
      b.q[1] = *(const uint4*)(Bl + 16);
      acc[nt] = __builtin_amdgcn_wmma_f32_16x16x32_bf16(
          false, a.v, false, b.v, (short)0, acc[nt], false, false);
    }
  }

  // 32-bit C/D layout: VGPR v -> row = base + v + 8*hi, col = lane%16
  #pragma unroll
  for (int nt = 0; nt < 4; ++nt) {
    const int col = panelN * 64 + nt * 16 + lm;
    #pragma unroll
    for (int v = 0; v < 8; ++v) {
      const int row = blockM + wv * 16 + hi * 8 + v;
      if (row < M) C[(size_t)row * Ncol + col] = acc[nt][v];
    }
  }
}

// ---------------------------------------------------------------------------
// Elementwise / conversion kernels
// ---------------------------------------------------------------------------
__global__ void fill_f32(float* __restrict__ p, float v, size_t n) {
  size_t i = (size_t)blockIdx.x * blockDim.x + threadIdx.x;
  if (i < n) p[i] = v;
}

__global__ void cvt_bf16(const float* __restrict__ in,
                         unsigned short* __restrict__ out, size_t n) {
  size_t i = (size_t)blockIdx.x * blockDim.x + threadIdx.x;
  if (i < n) out[i] = f2bf(in[i]);
}

// in: float [K][Ncol] row-major  ->  out: bf16 [Ncol][K]  (B transposed)
__global__ void cvtT_bf16(const float* __restrict__ in,
                          unsigned short* __restrict__ out, int K, int Ncol) {
  int i = blockIdx.x * blockDim.x + threadIdx.x;
  if (i < K * Ncol) {
    int k = i / Ncol, c = i - k * Ncol;
    out[(size_t)c * K + k] = f2bf(in[i]);
  }
}

// el/er: per (node, head) dot of z-head-slice with attention vectors
__global__ void attn_proj(const float* __restrict__ z,
                          const float* __restrict__ al,
                          const float* __restrict__ ar,
                          float* __restrict__ el, float* __restrict__ er, int Nn) {
  int i = blockIdx.x * blockDim.x + threadIdx.x;
  if (i >= Nn * 8) return;
  int n = i >> 3, h = i & 7;
  const float* zp = z + (size_t)n * 128 + h * 16;
  float sl = 0.f, sr = 0.f;
  #pragma unroll
  for (int d = 0; d < 16; ++d) { sl += zp[d] * al[h * 16 + d]; sr += zp[d] * ar[h * 16 + d]; }
  el[i] = sl; er[i] = sr;
}

__device__ __forceinline__ void atomicMaxFloat(float* addr, float v) {
  if (v >= 0.0f) atomicMax((int*)addr, __float_as_int(v));
  else           atomicMin((unsigned int*)addr, __float_as_uint(v));
}

__global__ void edge_max(const int* __restrict__ src, const int* __restrict__ dst,
                         const float* __restrict__ el, const float* __restrict__ er,
                         float* __restrict__ emax, int E) {
  size_t i = (size_t)blockIdx.x * blockDim.x + threadIdx.x;
  if (i >= (size_t)E * 8) return;
  int e = (int)(i >> 3), h = (int)(i & 7);
  int s = src[e], d = dst[e];
  float v = el[s * 8 + h] + er[d * 8 + h];
  v = v >= 0.f ? v : 0.2f * v;                 // leaky_relu(., 0.2)
  atomicMaxFloat(&emax[d * 8 + h], v);
}

__global__ void emax_fixup(float* __restrict__ emax, size_t n) {
  size_t i = (size_t)blockIdx.x * blockDim.x + threadIdx.x;
  if (i < n) { float v = emax[i]; if (isinf(v)) emax[i] = 0.0f; }  // zero in-degree guard
}

// 128 threads per edge: t = h*16 + d.  Coalesced gather of z[src], coalesced
// atomicAdd scatter into agg[dst] (all resident in L2).
__global__ void edge_scatter(const int* __restrict__ src, const int* __restrict__ dst,
                             const float* __restrict__ el, const float* __restrict__ er,
                             const float* __restrict__ emax, const float* __restrict__ z,
                             float* __restrict__ denom, float* __restrict__ agg, int E) {
  size_t i = (size_t)blockIdx.x * blockDim.x + threadIdx.x;
  if (i >= (size_t)E * 128) return;
  int e = (int)(i >> 7), t = (int)(i & 127);
  int h = t >> 4, d = t & 15;
  int s = src[e], dd = dst[e];
  float v = el[s * 8 + h] + er[dd * 8 + h];
  v = v >= 0.f ? v : 0.2f * v;
  float ex = __expf(v - emax[dd * 8 + h]);
  if (d == 0) atomicAdd(&denom[dd * 8 + h], ex);
  atomicAdd(&agg[(size_t)dd * 128 + t], ex * z[(size_t)s * 128 + t]);
}

__device__ __forceinline__ float blocksum128(float v) {
  __shared__ float red[128];
  int t = threadIdx.x;
  red[t] = v; __syncthreads();
  #pragma unroll
  for (int s = 64; s >= 1; s >>= 1) { if (t < s) red[t] += red[t + s]; __syncthreads(); }
  float r = red[0]; __syncthreads();
  return r;
}

// agg/denom -> +bias -> PReLU -> LN (h, f32) -> LN again (hn, bf16)
__global__ void finalize_node(const float* __restrict__ agg, const float* __restrict__ denom,
                              const float* __restrict__ cb, const float* __restrict__ a_conv, int l,
                              const float* __restrict__ lns, const float* __restrict__ lnb,
                              float* __restrict__ h_out, unsigned short* __restrict__ hnb) {
  const int n = blockIdx.x, t = threadIdx.x;
  float den = fmaxf(denom[n * 8 + (t >> 4)], 1e-9f);
  float v = agg[(size_t)n * 128 + t] / den + cb[t];
  float a = a_conv[l];
  v = v >= 0.f ? v : a * v;
  float m   = blocksum128(v) * (1.0f / 128.0f);
  float var = blocksum128((v - m) * (v - m)) * (1.0f / 128.0f);
  float hv  = (v - m) * rsqrtf(var + 1e-5f) * lns[t] + lnb[t];
  h_out[(size_t)n * 128 + t] = hv;
  float m2   = blocksum128(hv) * (1.0f / 128.0f);
  float var2 = blocksum128((hv - m2) * (hv - m2)) * (1.0f / 128.0f);
  float hn   = (hv - m2) * rsqrtf(var2 + 1e-5f) * lns[t] + lnb[t];
  hnb[(size_t)n * 128 + t] = f2bf(hn);
}

__global__ void ff1_act(const float* __restrict__ ffh, const float* __restrict__ b1,
                        const float* __restrict__ a_ff, int l,
                        unsigned short* __restrict__ ffhb, size_t n) {
  size_t i = (size_t)blockIdx.x * blockDim.x + threadIdx.x;
  if (i >= n) return;
  int c = (int)(i & 511);
  float a = a_ff[l];
  float v = ffh[i] + b1[c];
  v = v >= 0.f ? v : a * v;
  ffhb[i] = f2bf(v);
}

// out = h + ff2 + b2 ; also stage next-layer bf16 input
__global__ void ff2_add(const float* __restrict__ h, const float* __restrict__ ff2,
                        const float* __restrict__ b2, float* __restrict__ out,
                        unsigned short* __restrict__ xb, size_t n) {
  size_t i = (size_t)blockIdx.x * blockDim.x + threadIdx.x;
  if (i >= n) return;
  int c = (int)(i & 127);
  float v = h[i] + ff2[i] + b2[c];
  out[i] = v;
  xb[i] = f2bf(v);
}

// ---------------------------------------------------------------------------
// Host side
// ---------------------------------------------------------------------------
static inline unsigned cdiv(size_t a, size_t b) { return (unsigned)((a + b - 1) / b); }

extern "C" void kernel_launch(void* const* d_in, const int* in_sizes, int n_in,
                              void* d_out, int out_size, void* d_ws, size_t ws_size,
                              hipStream_t stream) {
  const float* in_feats  = (const float*)d_in[0];
  const int*   src       = (const int*)  d_in[1];
  const int*   dst       = (const int*)  d_in[2];
  const float* W         = (const float*)d_in[3];
  const float* attn_l    = (const float*)d_in[4];
  const float* attn_r    = (const float*)d_in[5];
  const float* conv_bias = (const float*)d_in[6];
  const float* a_conv    = (const float*)d_in[7];
  const float* ln_scale  = (const float*)d_in[8];
  const float* ln_bias   = (const float*)d_in[9];
  const float* W1        = (const float*)d_in[10];
  const float* b1        = (const float*)d_in[11];
  const float* W2        = (const float*)d_in[12];
  const float* b2        = (const float*)d_in[13];
  const float* a_ff      = (const float*)d_in[14];
  float* out = (float*)d_out;

  const int N = in_sizes[0] / 128;
  const int E = in_sizes[1];
  const int L = in_sizes[7];

  // bump allocator in workspace (~290 MB total)
  char* ws = (char*)d_ws;
  auto alloc = [&](size_t bytes) -> char* {
    char* p = ws; ws += (bytes + 255) & ~(size_t)255; return p;
  };
  float* z    = (float*)alloc((size_t)N * 128 * 4);
  float* agg  = (float*)alloc((size_t)N * 128 * 4);
  float* h    = (float*)alloc((size_t)N * 128 * 4);
  float* ff2  = (float*)alloc((size_t)N * 128 * 4);
  float* ffh  = (float*)alloc((size_t)N * 512 * 4);
  float* el   = (float*)alloc((size_t)N * 8 * 4);
  float* er   = (float*)alloc((size_t)N * 8 * 4);
  float* emax = (float*)alloc((size_t)N * 8 * 4);
  float* den  = (float*)alloc((size_t)N * 8 * 4);
  unsigned short* xb   = (unsigned short*)alloc((size_t)N * 128 * 2);
  unsigned short* hnb  = (unsigned short*)alloc((size_t)N * 128 * 2);
  unsigned short* ffhb = (unsigned short*)alloc((size_t)N * 512 * 2);
  unsigned short* Wt   = (unsigned short*)alloc((size_t)128 * 128 * 2);
  unsigned short* W1t  = (unsigned short*)alloc((size_t)128 * 512 * 2);
  unsigned short* W2t  = (unsigned short*)alloc((size_t)512 * 128 * 2);
  (void)ws_size; (void)n_in; (void)out_size;

  cvt_bf16<<<cdiv((size_t)N * 128, 256), 256, 0, stream>>>(in_feats, xb, (size_t)N * 128);

  for (int l = 0; l < L; ++l) {
    const float* Wl  = W  + (size_t)l * 128 * 128;
    const float* W1l = W1 + (size_t)l * 128 * 512;
    const float* W2l = W2 + (size_t)l * 512 * 128;
    const float* al  = attn_l    + (size_t)l * 128;
    const float* ar  = attn_r    + (size_t)l * 128;
    const float* cb  = conv_bias + (size_t)l * 128;
    const float* lns = ln_scale  + (size_t)l * 128;
    const float* lnb = ln_bias   + (size_t)l * 128;
    const float* b1l = b1 + (size_t)l * 512;
    const float* b2l = b2 + (size_t)l * 128;
    float* out_l = out + (size_t)l * N * 128;

    // convert + transpose weights to bf16 (tiny)
    cvtT_bf16<<<cdiv(128 * 128, 256), 256, 0, stream>>>(Wl,  Wt,  128, 128);
    cvtT_bf16<<<cdiv(128 * 512, 256), 256, 0, stream>>>(W1l, W1t, 128, 512);
    cvtT_bf16<<<cdiv(512 * 128, 256), 256, 0, stream>>>(W2l, W2t, 512, 128);

    // z = x @ W          [N,128] x [128,128]
    gemm_panel_wmma<128><<<dim3(cdiv(N, 128), 128 / 64), 256, 0, stream>>>(
        xb, Wt, z, N, 128);

    attn_proj<<<cdiv((size_t)N * 8, 256), 256, 0, stream>>>(z, al, ar, el, er, N);

    fill_f32<<<cdiv((size_t)N * 128, 256), 256, 0, stream>>>(agg, 0.f, (size_t)N * 128);
    fill_f32<<<cdiv((size_t)N * 8, 256), 256, 0, stream>>>(den, 0.f, (size_t)N * 8);
    fill_f32<<<cdiv((size_t)N * 8, 256), 256, 0, stream>>>(emax, -INFINITY, (size_t)N * 8);

    edge_max    <<<cdiv((size_t)E * 8,   256), 256, 0, stream>>>(src, dst, el, er, emax, E);
    emax_fixup  <<<cdiv((size_t)N * 8,   256), 256, 0, stream>>>(emax, (size_t)N * 8);
    edge_scatter<<<cdiv((size_t)E * 128, 256), 256, 0, stream>>>(src, dst, el, er, emax, z,
                                                                 den, agg, E);

    finalize_node<<<N, 128, 0, stream>>>(agg, den, cb, a_conv, l, lns, lnb, h, hnb);

    // ffh = hn @ W1      [N,128] x [128,512]
    gemm_panel_wmma<128><<<dim3(cdiv(N, 128), 512 / 64), 256, 0, stream>>>(
        hnb, W1t, ffh, N, 512);

    ff1_act<<<cdiv((size_t)N * 512, 256), 256, 0, stream>>>(ffh, b1l, a_ff, l, ffhb,
                                                            (size_t)N * 512);

    // ff2 = act @ W2     [N,512] x [512,128]
    gemm_panel_wmma<512><<<dim3(cdiv(N, 128), 128 / 64), 256, 0, stream>>>(
        ffhb, W2t, ff2, N, 128);

    ff2_add<<<cdiv((size_t)N * 128, 256), 256, 0, stream>>>(h, ff2, b2l, out_l, xb,
                                                            (size_t)N * 128);
  }
}
